// Block_27212912788319
// MI455X (gfx1250) — compile-verified, matrix operand order checked by reading, not run on hardware
//
#include <hip/hip_runtime.h>
#include <hip/hip_bf16.h>

#define Bz   2
#define Lz   2048
#define Hh   16
#define Dd   64
#define HIDz 1024
#define QKVPz 7168
#define VPz  5120
#define BLz  (Bz * Lz)

typedef __attribute__((ext_vector_type(16))) _Float16 v16h;
typedef __attribute__((ext_vector_type(8)))  _Float16 v8h;
typedef __attribute__((ext_vector_type(8)))  float    v8f;

// ---------------------------------------------------------------------------
// WMMA helper: D = A(16x32 f16) x B(32x16 f16) + C(16x16 f32)
// ---------------------------------------------------------------------------
__device__ __forceinline__ v8f wmma16(v16h a, v16h b, v8f c) {
  return __builtin_amdgcn_wmma_f32_16x16x32_f16(
      /*neg_a=*/false, a, /*neg_b=*/false, b,
      /*c_mod=*/(short)0, c, /*reuse_a=*/false, /*reuse_b=*/false);
}

// Fragment load from a row-major [rows x K] f16 matrix (global).
// Per ISA layout: lane<16 holds row r0+lane, K = {h8..h8+7, 16+h8..16+h8+7}
// with h8 = (lane>=16)*8 selecting the K sub-runs.
__device__ __forceinline__ v16h gfrag(const _Float16* __restrict__ base,
                                      int ld, int r0, int k0, int lane) {
  int half = (lane >> 4) & 1;
  const _Float16* p = base + (size_t)(r0 + (lane & 15)) * ld + k0 + half * 8;
  v8h lo = *reinterpret_cast<const v8h*>(p);
  v8h hi = *reinterpret_cast<const v8h*>(p + 16);
  v16h f;
#pragma unroll
  for (int i = 0; i < 8; ++i) { f[i] = lo[i]; f[i + 8] = hi[i]; }
  return f;
}

// Same fragment pattern but from an LDS tile with row stride ld (f16 elems).
__device__ __forceinline__ v16h lfrag(const _Float16* base, int ld, int lane) {
  int half = (lane >> 4) & 1;
  const _Float16* p = base + (lane & 15) * ld + half * 8;
  v8h lo = *reinterpret_cast<const v8h*>(p);
  v8h hi = *reinterpret_cast<const v8h*>(p + 16);
  v16h f;
#pragma unroll
  for (int i = 0; i < 8; ++i) { f[i] = lo[i]; f[i + 8] = hi[i]; }
  return f;
}

// ---------------------------------------------------------------------------
// f32 -> f16 convert
// ---------------------------------------------------------------------------
__global__ void f32_to_f16_kernel(const float* __restrict__ in,
                                  _Float16* __restrict__ out, int n) {
  int i = blockIdx.x * blockDim.x + threadIdx.x;
  if (i < n) out[i] = (_Float16)in[i];
}

// ---------------------------------------------------------------------------
// Block reduction (256 threads, 8 waves) for sum + sumsq
// ---------------------------------------------------------------------------
__device__ __forceinline__ void block_reduce2(float& s, float& s2, float* red) {
#pragma unroll
  for (int off = 16; off; off >>= 1) {
    s  += __shfl_xor(s,  off);
    s2 += __shfl_xor(s2, off);
  }
  int w = threadIdx.x >> 5;
  if ((threadIdx.x & 31) == 0) { red[w] = s; red[8 + w] = s2; }
  __syncthreads();
  s = 0.f; s2 = 0.f;
#pragma unroll
  for (int i = 0; i < 8; ++i) { s += red[i]; s2 += red[8 + i]; }
  __syncthreads();
}

// ---------------------------------------------------------------------------
// LayerNorm over 1024-wide rows. OutT = _Float16 (input LN) or float (final).
// One block per row, 256 threads, 4 elements/thread.
// ---------------------------------------------------------------------------
template <typename OutT>
__global__ __launch_bounds__(256) void ln1024_kernel(
    const float* __restrict__ in, const float* __restrict__ g,
    const float* __restrict__ bb, OutT* __restrict__ out) {
  __shared__ float red[16];
  int row = blockIdx.x;
  const float* r = in + (size_t)row * HIDz;
  float v[4], s = 0.f, s2 = 0.f;
#pragma unroll
  for (int i = 0; i < 4; ++i) {
    v[i] = r[threadIdx.x + i * 256];
    s += v[i]; s2 += v[i] * v[i];
  }
  block_reduce2(s, s2, red);
  float mean = s * (1.0f / HIDz);
  float rstd = rsqrtf(s2 * (1.0f / HIDz) - mean * mean + 1e-5f);
#pragma unroll
  for (int i = 0; i < 4; ++i) {
    int o = threadIdx.x + i * 256;
    float y = (v[i] - mean) * rstd * g[o] + bb[o];
    out[(size_t)row * HIDz + o] = (OutT)y;
  }
}

// ---------------------------------------------------------------------------
// Channel shift gather: A[b,l,c] = ln[b, l - sh(c), c] (zero pad before seq 0)
// ---------------------------------------------------------------------------
__global__ void shift_gather_kernel(const _Float16* __restrict__ ln,
                                    _Float16* __restrict__ a) {
  int idx = blockIdx.x * blockDim.x + threadIdx.x;
  if (idx >= BLz * HIDz) return;
  int c = idx & (HIDz - 1);
  int row = idx >> 10;
  int l = row & (Lz - 1);
  int b = row >> 11;
  int sl = (c < 128) ? l - 1 : (c < 256 ? l - 3 : l);
  _Float16 v = (_Float16)0.0f;
  if (sl >= 0) v = ln[((size_t)b * Lz + sl) * HIDz + c];
  a[idx] = v;
}

// ---------------------------------------------------------------------------
// WMMA GEMM: C[M,N] = A[M,K] * Bm[N,K]^T (both f16 row-major), C f16 or f32.
// Block tile 128x64, 8 waves; each wave -> 16 rows x 4 n-tiles.
// ---------------------------------------------------------------------------
template <bool HALF_OUT>
__global__ __launch_bounds__(256) void gemm_f16_kernel(
    const _Float16* __restrict__ A, const _Float16* __restrict__ Bm,
    void* __restrict__ C, int K, int ldc) {
  constexpr int LDP = 40;  // padded LDS row stride (f16) -> conflict-free
  __shared__ __align__(16) _Float16 As[128 * LDP];
  __shared__ __align__(16) _Float16 Bs[64 * LDP];
  const int bm = blockIdx.y * 128;
  const int bn = blockIdx.x * 64;
  const int t = threadIdx.x;
  const int lane = t & 31, wave = t >> 5;
  v8f acc[4] = {};

  for (int k0 = 0; k0 < K; k0 += 32) {
    // Stage A tile: 128 rows x 32 f16 (512 16B chunks / 256 threads)
#pragma unroll
    for (int i = 0; i < 2; ++i) {
      int c = t + i * 256;
      int r = c >> 2, cc = (c & 3) * 8;
      const v8h* src =
          reinterpret_cast<const v8h*>(A + (size_t)(bm + r) * K + k0 + cc);
      *reinterpret_cast<v8h*>(&As[r * LDP + cc]) = *src;
    }
    // Stage B tile: 64 rows (n) x 32 f16
    {
      int r = t >> 2, cc = (t & 3) * 8;
      const v8h* src =
          reinterpret_cast<const v8h*>(Bm + (size_t)(bn + r) * K + k0 + cc);
      *reinterpret_cast<v8h*>(&Bs[r * LDP + cc]) = *src;
    }
    __syncthreads();
    v16h af = lfrag(&As[(wave * 16) * LDP], LDP, lane);
#pragma unroll
    for (int nt = 0; nt < 4; ++nt) {
      v16h bf = lfrag(&Bs[(nt * 16) * LDP], LDP, lane);
      acc[nt] = wmma16(af, bf, acc[nt]);
    }
    __syncthreads();
  }
  // Epilogue. C layout: lane<16 -> n=lane, m=vgpr r; lane>=16 -> m=8+r.
  const int half = lane >> 4, nn = lane & 15;
#pragma unroll
  for (int nt = 0; nt < 4; ++nt) {
    int col = bn + nt * 16 + nn;
#pragma unroll
    for (int r = 0; r < 8; ++r) {
      int row = bm + wave * 16 + half * 8 + r;
      float v = acc[nt][r];
      if (HALF_OUT)
        ((_Float16*)C)[(size_t)row * ldc + col] = (_Float16)v;
      else
        ((float*)C)[(size_t)row * ldc + col] = v;
    }
  }
}

// ---------------------------------------------------------------------------
// Mid: relu + LN(7168) + split into q,k (B,H,L,D), vT (B,H,D,L) and
// gelu(p)*roll(p,1) written into concat[:,1024:5120]. One block per token row.
// ---------------------------------------------------------------------------
__global__ __launch_bounds__(256) void mid_kernel(
    const _Float16* __restrict__ hpre, const float* __restrict__ g,
    const float* __restrict__ bb, _Float16* __restrict__ qb,
    _Float16* __restrict__ kb, _Float16* __restrict__ vtb,
    _Float16* __restrict__ catb) {
  __shared__ float psh[4096];
  __shared__ float red[16];
  const int row = blockIdx.x;
  const int l = row & (Lz - 1);
  const int b = row >> 11;
  const _Float16* hr = hpre + (size_t)row * QKVPz;
  float vals[28], s = 0.f, s2 = 0.f;
#pragma unroll
  for (int i = 0; i < 28; ++i) {
    float v = (float)hr[threadIdx.x + i * 256];
    v = v > 0.f ? v : 0.f;  // relu before LN
    vals[i] = v; s += v; s2 += v * v;
  }
  block_reduce2(s, s2, red);
  const float mean = s * (1.0f / QKVPz);
  const float rstd = rsqrtf(s2 * (1.0f / QKVPz) - mean * mean + 1e-5f);
#pragma unroll
  for (int i = 0; i < 28; ++i) {
    int o = threadIdx.x + i * 256;
    float y = (vals[i] - mean) * rstd * g[o] + bb[o];
    if (o < 1024) {
      int h = o >> 6, d = o & 63;
      qb[(((size_t)b * Hh + h) * Lz + l) * Dd + d] = (_Float16)y;
    } else if (o < 2048) {
      int oo = o - 1024; int h = oo >> 6, d = oo & 63;
      kb[(((size_t)b * Hh + h) * Lz + l) * Dd + d] = (_Float16)y;
    } else if (o < 3072) {
      int oo = o - 2048; int h = oo >> 6, d = oo & 63;
      vtb[(((size_t)b * Hh + h) * Dd + d) * Lz + l] = (_Float16)y;  // transposed
    } else {
      psh[o - 3072] = y;
    }
  }
  __syncthreads();
#pragma unroll
  for (int i = 0; i < 16; ++i) {
    int j = threadIdx.x + i * 256;
    float p = psh[j];
    float pm = psh[(j + 4095) & 4095];  // roll(p, 1)
    float ge = 0.5f * p * (1.0f + erff(p * 0.70710678f));
    catb[(size_t)row * VPz + HIDz + j] = (_Float16)(ge * pm);
  }
}

// ---------------------------------------------------------------------------
// Flash attention: 1 wave per (b,h,i-tile of 16 rows). j-step of 32 keys.
// scores = (Q K^T)/8 + j*slope, causal; online softmax; O = P V.
// Output (scaled 1/l) to concat[:, h*64 .. h*64+63].
// ---------------------------------------------------------------------------
__global__ __launch_bounds__(32) void attn_kernel(
    const _Float16* __restrict__ qb, const _Float16* __restrict__ kb,
    const _Float16* __restrict__ vtb, const float* __restrict__ slopes,
    _Float16* __restrict__ catb) {
  __shared__ __align__(16) _Float16 psh[16 * 40];
  const int blk = blockIdx.x;
  const int it = blk & (Lz / 16 - 1);
  const int bh = blk >> 7;
  const int h = bh & (Hh - 1);
  const int b = bh >> 4;
  const int i0 = it * 16;
  const int lane = threadIdx.x;
  const int half = lane >> 4, lr = lane & 15;
  const _Float16* Q  = qb  + (size_t)bh * Lz * Dd;
  const _Float16* Kp = kb  + (size_t)bh * Lz * Dd;
  const _Float16* Vt = vtb + (size_t)bh * Dd * Lz;
  const float slope = slopes[h];
  const float scale = 0.125f;  // D^-0.5

  const v16h aq0 = gfrag(Q, Dd, i0, 0, lane);
  const v16h aq1 = gfrag(Q, Dd, i0, 32, lane);
  v8f o0 = {}, o1 = {}, o2 = {}, o3 = {};
  float mst[8], lst[8];
#pragma unroll
  for (int r = 0; r < 8; ++r) { mst[r] = -1e30f; lst[r] = 0.f; }

  for (int j0 = 0; j0 <= i0 + 15; j0 += 32) {
    // Score tiles: S0 (keys j0..j0+15), S1 (j0+16..j0+31), K-dim = D = 64.
    v8f s0 = {}, s1 = {};
    s0 = wmma16(aq0, gfrag(Kp, Dd, j0, 0, lane), s0);
    s0 = wmma16(aq1, gfrag(Kp, Dd, j0, 32, lane), s0);
    s1 = wmma16(aq0, gfrag(Kp, Dd, j0 + 16, 0, lane), s1);
    s1 = wmma16(aq1, gfrag(Kp, Dd, j0 + 16, 32, lane), s1);

    float p0[8], p1[8];
#pragma unroll
    for (int r = 0; r < 8; ++r) {
      int m = i0 + half * 8 + r;
      int n0 = j0 + lr, n1 = j0 + 16 + lr;
      float v0 = s0[r] * scale + slope * (float)n0;
      float v1 = s1[r] * scale + slope * (float)n1;
      if (n0 > m) v0 = -1e9f;
      if (n1 > m) v1 = -1e9f;
      float mx = fmaxf(v0, v1);
#pragma unroll
      for (int off = 8; off; off >>= 1) mx = fmaxf(mx, __shfl_xor(mx, off));
      float nm = fmaxf(mst[r], mx);
      float f = __expf(mst[r] - nm);
      v0 = __expf(v0 - nm);
      v1 = __expf(v1 - nm);
      float rs = v0 + v1;
#pragma unroll
      for (int off = 8; off; off >>= 1) rs += __shfl_xor(rs, off);
      lst[r] = lst[r] * f + rs;
      mst[r] = nm;
      o0[r] *= f; o1[r] *= f; o2[r] *= f; o3[r] *= f;
      p0[r] = v0; p1[r] = v1;
    }
    // Re-layout P (C-layout) -> A-fragment via LDS (16 x 32, stride 40).
#pragma unroll
    for (int r = 0; r < 8; ++r) {
      int m = half * 8 + r;
      psh[m * 40 + lr] = (_Float16)p0[r];
      psh[m * 40 + 16 + lr] = (_Float16)p1[r];
    }
    asm volatile("s_wait_dscnt 0" ::: "memory");
    v16h ap = lfrag(psh, 40, lane);
    // O += P * V : B[k=j(32), n=d(16)] from V^T rows d, contiguous in j.
    o0 = wmma16(ap, gfrag(Vt, Lz, 0,  j0, lane), o0);
    o1 = wmma16(ap, gfrag(Vt, Lz, 16, j0, lane), o1);
    o2 = wmma16(ap, gfrag(Vt, Lz, 32, j0, lane), o2);
    o3 = wmma16(ap, gfrag(Vt, Lz, 48, j0, lane), o3);
    asm volatile("s_wait_dscnt 0" ::: "memory");
  }
  // Epilogue: divide by row sums, write into concat[:, h*64 + d].
#pragma unroll
  for (int r = 0; r < 8; ++r) {
    int m = i0 + half * 8 + r;
    float inv = 1.0f / lst[r];
    size_t ro = ((size_t)b * Lz + m) * VPz + h * 64;
    catb[ro + 0  + lr] = (_Float16)(o0[r] * inv);
    catb[ro + 16 + lr] = (_Float16)(o1[r] * inv);
    catb[ro + 32 + lr] = (_Float16)(o2[r] * inv);
    catb[ro + 48 + lr] = (_Float16)(o3[r] * inv);
  }
}

// ---------------------------------------------------------------------------
// Host launcher
// ---------------------------------------------------------------------------
extern "C" void kernel_launch(void* const* d_in, const int* in_sizes, int n_in,
                              void* d_out, int out_size, void* d_ws,
                              size_t ws_size, hipStream_t stream) {
  const float* x        = (const float*)d_in[0];
  const float* in_ln_g  = (const float*)d_in[1];
  const float* in_ln_b  = (const float*)d_in[2];
  const float* W_in     = (const float*)d_in[3];
  const float* mid_ln_g = (const float*)d_in[4];
  const float* mid_ln_b = (const float*)d_in[5];
  const float* slopes   = (const float*)d_in[6];
  const float* W_out    = (const float*)d_in[7];
  const float* out_ln_g = (const float*)d_in[8];
  const float* out_ln_b = (const float*)d_in[9];
  float* out = (float*)d_out;

  size_t off = 0;
  auto carve = [&](size_t bytes) {
    void* p = (char*)d_ws + off;
    off += (bytes + 255) & ~(size_t)255;
    return p;
  };
  _Float16* wInH  = (_Float16*)carve((size_t)QKVPz * HIDz * 2);
  _Float16* wOutH = (_Float16*)carve((size_t)HIDz * VPz * 2);
  _Float16* lnH   = (_Float16*)carve((size_t)BLz * HIDz * 2);
  _Float16* aH    = (_Float16*)carve((size_t)BLz * HIDz * 2);
  _Float16* hpreH = (_Float16*)carve((size_t)BLz * QKVPz * 2);
  _Float16* qH    = (_Float16*)carve((size_t)BLz * HIDz * 2);
  _Float16* kH    = (_Float16*)carve((size_t)BLz * HIDz * 2);
  _Float16* vtH   = (_Float16*)carve((size_t)BLz * HIDz * 2);
  _Float16* catH  = (_Float16*)carve((size_t)BLz * VPz * 2);
  float*    opre  = (float*)carve((size_t)BLz * HIDz * 4);
  (void)ws_size; (void)n_in; (void)in_sizes; (void)out_size;

  // 1) weight conversion f32 -> f16
  {
    int n1 = QKVPz * HIDz;
    f32_to_f16_kernel<<<(n1 + 255) / 256, 256, 0, stream>>>(W_in, wInH, n1);
    int n2 = HIDz * VPz;
    f32_to_f16_kernel<<<(n2 + 255) / 256, 256, 0, stream>>>(W_out, wOutH, n2);
  }
  // 2) input layernorm
  ln1024_kernel<_Float16><<<BLz, 256, 0, stream>>>(x, in_ln_g, in_ln_b, lnH);
  // 3) channel shift -> GEMM1 A matrix
  shift_gather_kernel<<<(BLz * HIDz + 255) / 256, 256, 0, stream>>>(lnH, aH);
  // 4) GEMM1: hpre[4096,7168] = A[4096,1024] x W_in[7168,1024]^T
  gemm_f16_kernel<true><<<dim3(QKVPz / 64, BLz / 128), 256, 0, stream>>>(
      aH, wInH, hpreH, HIDz, QKVPz);
  // 5) relu + mid LN + split + gelu-gate
  mid_kernel<<<BLz, 256, 0, stream>>>(hpreH, mid_ln_g, mid_ln_b, qH, kH, vtH,
                                      catH);
  // 6) attention -> concat[:, 0:1024]
  attn_kernel<<<Bz * Hh * (Lz / 16), 32, 0, stream>>>(qH, kH, vtH, slopes,
                                                      catH);
  // 7) GEMM2: opre[4096,1024] = concat[4096,5120] x W_out[1024,5120]^T
  gemm_f16_kernel<false><<<dim3(HIDz / 64, BLz / 128), 256, 0, stream>>>(
      catH, wOutH, opre, VPz, HIDz);
  // 8) final layernorm
  ln1024_kernel<float><<<BLz, 256, 0, stream>>>(opre, out_ln_g, out_ln_b, out);
}
